// TextRNN_80135499808838
// MI455X (gfx1250) — matrix operand order repaired and checked by reading, not compiled, not run on hardware
//
#include <hip/hip_runtime.h>

// TextRNN on MI455X (gfx1250): persistent 16-WG grid, bf16 WMMA GEMMs with
// f32 accumulation. Weights pre-converted to bf16 (11.5 MB, L2-resident);
// embedding gather pre-staged to bf16 [t][m][k]. Each wave owns TWO 16x16
// output tiles (independent accumulators sharing one A fragment). Weight
// pointers are deliberately NOT noalias so the compiler cannot hoist the
// B-fragment loads out of the 512-step serial loop (that caused massive
// scratch spills); they re-load from L2 every step instead.

#define VOCAB 32000
#define EMB   512
#define HID   1024
#define NCLS  20
#define BATCH 64
#define SEQ   512
#define K1    (EMB + HID)   // 1536
#define K2    (HID + HID)   // 2048
#define NWG   16
#define NTHR  256

typedef __attribute__((ext_vector_type(8)))  float    v8f;
typedef __attribute__((ext_vector_type(16))) __bf16   v16bf;
typedef __attribute__((ext_vector_type(4)))  unsigned u32x4;

union Frag { v16bf v; u32x4 q[2]; unsigned short s[16]; };

__device__ __forceinline__ unsigned short f2bf(float f) {
  unsigned u = __float_as_uint(f);                       // RNE fp32 -> bf16
  return (unsigned short)((u + 0x7FFFu + ((u >> 16) & 1u)) >> 16);
}
__device__ __forceinline__ float bf2f(unsigned short h) {
  return __uint_as_float(((unsigned)h) << 16);
}

// A fragment (16x32 bf16): lane M = lane&15; 16 contiguous K starting at
// K = (lane>>4)*16 (matches B fragment order). Two global_load_b128.
__device__ __forceinline__ v16bf load_fragA_bf(const unsigned short* p) {
  Frag f;
  f.q[0] = *(const u32x4*)(p);
  f.q[1] = *(const u32x4*)(p + 8);
  return f.v;
}
// B fragment (32x16 bf16) from row-major W[N][K]: lane N = lane&15, lanes
// 16-31 take the upper K half; 16 contiguous bf16 = 32B (2x b128) load.
__device__ __forceinline__ v16bf load_fragB_bf(const unsigned short* p) {
  Frag f;
  f.q[0] = *(const u32x4*)(p);
  f.q[1] = *(const u32x4*)(p + 8);
  return f.v;
}

__device__ __forceinline__ void grid_bar(unsigned* cnt, unsigned target) {
  __threadfence();
  __syncthreads();
  if (threadIdx.x == 0) {
    __hip_atomic_fetch_add(cnt, 1u, __ATOMIC_ACQ_REL, __HIP_MEMORY_SCOPE_AGENT);
    while (__hip_atomic_load(cnt, __ATOMIC_RELAXED, __HIP_MEMORY_SCOPE_AGENT) < target)
      __builtin_amdgcn_s_sleep(2);
  }
  __syncthreads();
  // Per-wave acquire so this WGP's caches drop stale activation lines.
  (void)__hip_atomic_load(cnt, __ATOMIC_ACQUIRE, __HIP_MEMORY_SCOPE_AGENT);
  // Compiler-level barrier: nothing memory-carried may be hoisted across.
  asm volatile("" ::: "memory");
}

__global__ void cvt_f32_bf16(const float* __restrict__ src,
                             unsigned short* __restrict__ dst, int n) {
  int i = blockIdx.x * blockDim.x + threadIdx.x;
  int stride = gridDim.x * blockDim.x;
  for (; i < n; i += stride) dst[i] = f2bf(src[i]);
}

// One-time embedding gather + bf16 conversion: xe[t][m][k] = bf16(emb[x[m][t]][k]).
__global__ void gather_emb_bf16(const int* __restrict__ x,
                                const float* __restrict__ emb,
                                unsigned short* __restrict__ xe) {
  const long long nchunks = (long long)BATCH * SEQ * (EMB / 8);
  long long c = (long long)blockIdx.x * blockDim.x + threadIdx.x;
  const long long stride = (long long)gridDim.x * blockDim.x;
  for (; c < nchunks; c += stride) {
    int k   = ((int)c & 63) << 3;       // chunk-in-row * 8
    int row = (int)(c >> 6);            // row = t*BATCH + m
    int m   = row & (BATCH - 1);
    int t   = row >> 6;
    const float* src = emb + (size_t)x[m * SEQ + t] * EMB + k;
    unsigned short* dst = xe + ((size_t)t * BATCH + m) * EMB + k;
#pragma unroll
    for (int j = 0; j < 8; ++j) dst[j] = f2bf(src[j]);
  }
}

// Dual-tile GEMM segment: acc{0,1} += A[0..klen) x W^T tiles. Partial unroll
// keeps VGPR pressure bounded while still folding offsets into immediates.
__device__ __forceinline__ void gemm_seg(const unsigned short* aRowP,
                                         const unsigned short* wR0,
                                         const unsigned short* wR1,
                                         int klen, int kqA, int kqB,
                                         v8f& acc0, v8f& acc1) {
#pragma clang loop unroll_count(8)
  for (int kb = 0; kb < klen; kb += 32) {
    v16bf a  = load_fragA_bf(aRowP + kb + kqB);
    v16bf b0 = load_fragB_bf(wR0 + kb + kqB);
    v16bf b1 = load_fragB_bf(wR1 + kb + kqB);
    acc0 = __builtin_amdgcn_wmma_f32_16x16x32_bf16(false, a, false, b0,
                                                   (short)0, acc0, false, false);
    acc1 = __builtin_amdgcn_wmma_f32_16x16x32_bf16(false, a, false, b1,
                                                   (short)0, acc1, false, false);
  }
}

// NOTE: weight/state pointers intentionally lack __restrict__ and const-ness
// guarantees strong enough for LICM across the time loop (see header comment).
__global__ void __launch_bounds__(NTHR, 1)
textrnn_persistent(const unsigned short* __restrict__ xe,
                   const unsigned short* W1i, const float* b1i,
                   const unsigned short* W1o, const float* b1o,
                   const unsigned short* W2i, const float* b2i,
                   const unsigned short* W2o, const float* b2o,
                   const float* Wfc, const float* bfc,
                   unsigned short* h1, unsigned short* h2,
                   unsigned short* t1, unsigned short* t2,
                   unsigned* cnt, float* __restrict__ out) {
  const int tid  = threadIdx.x;
  const int lane = tid & 31;
  const int wave = tid >> 5;
  const int m0   = (wave & 3) << 4;                             // 4 M-tiles
  const int n0   = ((int)blockIdx.x << 6) + ((wave >> 2) << 5); // 2 N-tiles/wave
  const int r16  = lane & 15;
  const int hi   = lane >> 4;
  const int kqA  = hi << 3;   // M-group offset (0/8) in C/D
  const int kqB  = hi << 4;   // B K-half offset (0/16)
  const int aRow = m0 + r16;

  unsigned bar = 0;

  for (int t = 0; t < SEQ; ++t) {
    const unsigned short* h1r = h1 + (size_t)(t & 1) * (BATCH * HID);
    const unsigned short* h2r = h2 + (size_t)(t & 1) * (BATCH * HID);
    unsigned short* h1w = h1 + (size_t)((t + 1) & 1) * (BATCH * HID);
    unsigned short* h2w = h2 + (size_t)((t + 1) & 1) * (BATCH * HID);

    { // GEMM1a: t1 = tanh([xe_t | h1] @ W1i^T + b1i), K = 512 + 1024
      const unsigned short* xeRow = xe + ((size_t)t * BATCH + aRow) * EMB;
      const unsigned short* wR0 = W1i + (size_t)(n0 + r16) * K1;
      const unsigned short* wR1 = wR0 + (size_t)16 * K1;
      v8f acc0 = {}, acc1 = {};
      gemm_seg(xeRow,            wR0,       wR1,       EMB, kqA, kqB, acc0, acc1);
      gemm_seg(h1r + aRow * HID, wR0 + EMB, wR1 + EMB, HID, kqA, kqB, acc0, acc1);
      const float bv0 = b1i[n0 + r16], bv1 = b1i[n0 + 16 + r16];
#pragma unroll
      for (int r = 0; r < 8; ++r) {
        unsigned short* d = t1 + (m0 + kqA + r) * HID + n0 + r16;
        d[0]  = f2bf(tanhf(acc0[r] + bv0));
        d[16] = f2bf(tanhf(acc1[r] + bv1));
      }
    }
    grid_bar(cnt, ++bar * NWG);

    { // GEMM1b: h1w = t1 @ W1o^T + b1o, K = 1024
      const unsigned short* wR0 = W1o + (size_t)(n0 + r16) * HID;
      const unsigned short* wR1 = wR0 + (size_t)16 * HID;
      v8f acc0 = {}, acc1 = {};
      gemm_seg(t1 + aRow * HID, wR0, wR1, HID, kqA, kqB, acc0, acc1);
      const float bv0 = b1o[n0 + r16], bv1 = b1o[n0 + 16 + r16];
#pragma unroll
      for (int r = 0; r < 8; ++r) {
        unsigned short* d = h1w + (m0 + kqA + r) * HID + n0 + r16;
        d[0]  = f2bf(acc0[r] + bv0);
        d[16] = f2bf(acc1[r] + bv1);
      }
    }
    grid_bar(cnt, ++bar * NWG);

    { // GEMM2a: t2 = tanh([h1w | h2r] @ W2i^T + b2i), K = 1024 + 1024
      const unsigned short* wR0 = W2i + (size_t)(n0 + r16) * K2;
      const unsigned short* wR1 = wR0 + (size_t)16 * K2;
      v8f acc0 = {}, acc1 = {};
      gemm_seg(h1w + aRow * HID, wR0,       wR1,       HID, kqA, kqB, acc0, acc1);
      gemm_seg(h2r + aRow * HID, wR0 + HID, wR1 + HID, HID, kqA, kqB, acc0, acc1);
      const float bv0 = b2i[n0 + r16], bv1 = b2i[n0 + 16 + r16];
#pragma unroll
      for (int r = 0; r < 8; ++r) {
        unsigned short* d = t2 + (m0 + kqA + r) * HID + n0 + r16;
        d[0]  = f2bf(tanhf(acc0[r] + bv0));
        d[16] = f2bf(tanhf(acc1[r] + bv1));
      }
    }
    grid_bar(cnt, ++bar * NWG);

    { // GEMM2b: h2w = t2 @ W2o^T + b2o, K = 1024
      const unsigned short* wR0 = W2o + (size_t)(n0 + r16) * HID;
      const unsigned short* wR1 = wR0 + (size_t)16 * HID;
      v8f acc0 = {}, acc1 = {};
      gemm_seg(t2 + aRow * HID, wR0, wR1, HID, kqA, kqB, acc0, acc1);
      const float bv0 = b2o[n0 + r16], bv1 = b2o[n0 + 16 + r16];
#pragma unroll
      for (int r = 0; r < 8; ++r) {
        unsigned short* d = h2w + (m0 + kqA + r) * HID + n0 + r16;
        d[0]  = f2bf(acc0[r] + bv0);
        d[16] = f2bf(acc1[r] + bv1);
      }
    }
    grid_bar(cnt, ++bar * NWG);
  }

  // Classifier: out = h2 @ Wfc^T + bfc  (final h2 in parity 0; 64x20, one-time)
  if (blockIdx.x == 0) {
    const unsigned short* hf = h2;  // SEQ even -> parity 0
    for (int i = tid; i < BATCH * NCLS; i += NTHR) {
      int m = i / NCLS, c = i % NCLS;
      float s = bfc[c];
      const unsigned short* hrow = hf + m * HID;
      const float* wrow = Wfc + (size_t)c * HID;
      for (int k = 0; k < HID; ++k) s += bf2f(hrow[k]) * wrow[k];
      out[i] = s;
    }
  }
}

extern "C" void kernel_launch(void* const* d_in, const int* in_sizes, int n_in,
                              void* d_out, int out_size, void* d_ws, size_t ws_size,
                              hipStream_t stream) {
  const int*   x    = (const int*)d_in[0];
  const float* emb  = (const float*)d_in[1];
  const float* W1i  = (const float*)d_in[2];
  const float* b1i  = (const float*)d_in[3];
  const float* W1o  = (const float*)d_in[4];
  const float* b1o  = (const float*)d_in[5];
  const float* W2i  = (const float*)d_in[6];
  const float* b2i  = (const float*)d_in[7];
  const float* W2o  = (const float*)d_in[8];
  const float* b2o  = (const float*)d_in[9];
  const float* Wfc  = (const float*)d_in[10];
  const float* bfc  = (const float*)d_in[11];

  char* ws = (char*)d_ws;
  size_t off = 0;
  auto alloc = [&](size_t bytes) -> char* {
    char* p = ws + off;
    off = (off + bytes + 255) & ~(size_t)255;
    return p;
  };

  unsigned*       cnt    = (unsigned*)alloc(256);
  unsigned short* W1i_b  = (unsigned short*)alloc((size_t)HID * K1 * 2);
  unsigned short* W1o_b  = (unsigned short*)alloc((size_t)HID * HID * 2);
  unsigned short* W2i_b  = (unsigned short*)alloc((size_t)HID * K2 * 2);
  unsigned short* W2o_b  = (unsigned short*)alloc((size_t)HID * HID * 2);
  unsigned short* h1_b   = (unsigned short*)alloc((size_t)2 * BATCH * HID * 2);
  unsigned short* h2_b   = (unsigned short*)alloc((size_t)2 * BATCH * HID * 2);
  unsigned short* t1_b   = (unsigned short*)alloc((size_t)BATCH * HID * 2);
  unsigned short* t2_b   = (unsigned short*)alloc((size_t)BATCH * HID * 2);
  unsigned short* xe_b   = (unsigned short*)alloc((size_t)SEQ * BATCH * EMB * 2); // 32 MB

  hipMemsetAsync(cnt, 0, 256, stream);
  hipMemsetAsync(h1_b, 0, (size_t)2 * BATCH * HID * 2, stream);  // h0 = 0 (bf16 zero)
  hipMemsetAsync(h2_b, 0, (size_t)2 * BATCH * HID * 2, stream);

  cvt_f32_bf16<<<1024, 256, 0, stream>>>(W1i, W1i_b, HID * K1);
  cvt_f32_bf16<<<1024, 256, 0, stream>>>(W1o, W1o_b, HID * HID);
  cvt_f32_bf16<<<1024, 256, 0, stream>>>(W2i, W2i_b, HID * K2);
  cvt_f32_bf16<<<1024, 256, 0, stream>>>(W2o, W2o_b, HID * HID);
  gather_emb_bf16<<<4096, 256, 0, stream>>>(x, emb, xe_b);

  textrnn_persistent<<<NWG, NTHR, 0, stream>>>(
      xe_b, W1i_b, b1i, W1o_b, b1o, W2i_b, b2i, W2o_b, b2o, Wfc, bfc,
      h1_b, h2_b, t1_b, t2_b, cnt, (float*)d_out);
}
